// MLA_22548578304474
// MI455X (gfx1250) — compile-verified, matrix operand order checked by reading, not run on hardware
//
#include <hip/hip_runtime.h>
#include <hip/hip_bf16.h>

typedef __attribute__((ext_vector_type(16))) __bf16 v16bf;
typedef __attribute__((ext_vector_type(8)))  float  v8f;

#define B_   2
#define S_   2048
#define D_   2048
#define H_   16
#define QL_  1536
#define KVL_ 512
#define DN_  128
#define DR_  64
#define DV_  128
#define DQK_ 192
#define NTOK (B_*S_)
#define SM_SCALE 0.0721687836487032f   // 1/sqrt(192), folded into q

static __device__ __forceinline__ v8f wmma_bf16(v16bf a, v16bf b, v8f c) {
  return __builtin_amdgcn_wmma_f32_16x16x32_bf16(false, a, false, b, (short)0, c, false, false);
}

// Load a 16x32 bf16 fragment (A-layout / B-as-row-major layout) from a padded
// row-major tile. Lane l holds row (row0 + l%16); K base = 8*(l/16); elements
// 0..7 = K..K+7, 8..15 = K+16..K+23  -> two 16B loads.
static __device__ __forceinline__ v16bf ld_frag(const __bf16* tile, int row0, int col0, int ld, int lane) {
  const __bf16* p = tile + (size_t)(row0 + (lane & 15)) * ld + col0 + ((lane >> 4) << 3);
  v16bf f;
  ((uint4*)&f)[0] = *(const uint4*)(p);
  ((uint4*)&f)[1] = *(const uint4*)(p + 16);
  return f;
}

// ---------------------------------------------------------------- conversions
__global__ __launch_bounds__(256) void cvt_f32_bf16(const float* __restrict__ in,
                                                    __bf16* __restrict__ out, size_t n) {
  size_t i = (size_t)blockIdx.x * blockDim.x + threadIdx.x;
  size_t stride = (size_t)gridDim.x * blockDim.x;
  for (; i < n; i += stride) out[i] = (__bf16)in[i];
}

// ---------------------------------------------------------------- GEMM  C = A * B^T
// A: [M,K] bf16 row-major, B: [N,K] bf16 row-major.
// Block tile 256x128, BK=32, double-buffered LDS, 8 waves in 4x2, 64x64/wave.
#define BM 256
#define BN 128
#define BK 32
#define LDK 40   // padded lds stride (bf16 elems); 80B rows keep 16B alignment

template<bool OUT_BF16>
__global__ __launch_bounds__(256) void gemm_nt(const __bf16* __restrict__ A,
                                               const __bf16* __restrict__ Bm,
                                               float* __restrict__ Cf,
                                               __bf16* __restrict__ Cb,
                                               int M, int N, int K) {
  __shared__ __bf16 As[2][BM * LDK];   // 2 x 20.0 KB
  __shared__ __bf16 Bs[2][BN * LDK];   // 2 x 10.0 KB
  const int tid  = threadIdx.x;
  const int wave = tid >> 5, lane = tid & 31;
  const int wm = wave >> 1, wn = wave & 1;          // 4 x 2 wave grid
  const int block_m = blockIdx.y * BM;
  const int block_n = blockIdx.x * BN;
  const int half = lane >> 4, ln = lane & 15;

  // prologue: fill buffer 0 straight from global
  #pragma unroll
  for (int t = 0; t < 4; ++t) {
    int c = tid + t * 256;
    int r = c >> 2, col = (c & 3) << 3;
    uint4 v = make_uint4(0, 0, 0, 0);
    int gm = block_m + r;
    if (gm < M) v = *(const uint4*)(A + (size_t)gm * K + col);
    *(uint4*)(&As[0][r * LDK + col]) = v;
  }
  #pragma unroll
  for (int t = 0; t < 2; ++t) {
    int c = tid + t * 256;
    int r = c >> 2, col = (c & 3) << 3;
    uint4 v = make_uint4(0, 0, 0, 0);
    int gn = block_n + r;
    if (gn < N) v = *(const uint4*)(Bm + (size_t)gn * K + col);
    *(uint4*)(&Bs[0][r * LDK + col]) = v;
  }
  __syncthreads();

  v8f acc[4][4] = {};
  int buf = 0;
  for (int k0 = 0; k0 < K; k0 += BK) {
    const bool nxt = (k0 + BK) < K;
    uint4 pa[4], pb[2];
    if (nxt) {
      const int kn = k0 + BK;
      if (kn + BK < K) {   // software prefetch two tiles ahead -> global_prefetch
        __builtin_prefetch(A + (size_t)(block_m + (tid >> 2)) * K + kn + BK, 0, 0);
        __builtin_prefetch(Bm + (size_t)(block_n + (tid >> 3)) * K + kn + BK, 0, 0);
      }
      #pragma unroll
      for (int t = 0; t < 4; ++t) {
        int c = tid + t * 256;
        int r = c >> 2, col = (c & 3) << 3;
        int gm = block_m + r;
        pa[t] = make_uint4(0, 0, 0, 0);
        if (gm < M) pa[t] = *(const uint4*)(A + (size_t)gm * K + kn + col);
      }
      #pragma unroll
      for (int t = 0; t < 2; ++t) {
        int c = tid + t * 256;
        int r = c >> 2, col = (c & 3) << 3;
        int gn = block_n + r;
        pb[t] = make_uint4(0, 0, 0, 0);
        if (gn < N) pb[t] = *(const uint4*)(Bm + (size_t)gn * K + kn + col);
      }
    }

    // all fragments into distinct regs first, then the WMMA burst
    v16bf af[4], bfr[4];
    #pragma unroll
    for (int i = 0; i < 4; ++i) af[i]  = ld_frag(&As[buf][0], wm * 64 + i * 16, 0, LDK, lane);
    #pragma unroll
    for (int j = 0; j < 4; ++j) bfr[j] = ld_frag(&Bs[buf][0], wn * 64 + j * 16, 0, LDK, lane);
    #pragma unroll
    for (int i = 0; i < 4; ++i)
      #pragma unroll
      for (int j = 0; j < 4; ++j)
        acc[i][j] = wmma_bf16(af[i], bfr[j], acc[i][j]);

    if (nxt) {
      #pragma unroll
      for (int t = 0; t < 4; ++t) {
        int c = tid + t * 256;
        int r = c >> 2, col = (c & 3) << 3;
        *(uint4*)(&As[buf ^ 1][r * LDK + col]) = pa[t];
      }
      #pragma unroll
      for (int t = 0; t < 2; ++t) {
        int c = tid + t * 256;
        int r = c >> 2, col = (c & 3) << 3;
        *(uint4*)(&Bs[buf ^ 1][r * LDK + col]) = pb[t];
      }
    }
    __syncthreads();
    buf ^= 1;
  }

  #pragma unroll
  for (int i = 0; i < 4; ++i)
    #pragma unroll
    for (int j = 0; j < 4; ++j) {
      int nn = block_n + wn * 64 + j * 16 + ln;
      if (nn < N) {
        int mbase = block_m + wm * 64 + i * 16 + half * 8;
        #pragma unroll
        for (int r = 0; r < 8; ++r) {
          int mm = mbase + r;
          if (mm < M) {
            float v = acc[i][j][r];
            if (OUT_BF16) Cb[(size_t)mm * N + nn] = (__bf16)v;
            else          Cf[(size_t)mm * N + nn] = v;
          }
        }
      }
    }
}

// ---------------------------------------------------------------- RMSNorm rows
__global__ __launch_bounds__(256) void rmsnorm_rows(const float* __restrict__ in,
                                                    const float* __restrict__ w,
                                                    __bf16* __restrict__ out, int ncols) {
  const int row = blockIdx.x;
  const float* x = in + (size_t)row * ncols;
  float s = 0.f;
  for (int c = threadIdx.x; c < ncols; c += 256) { float v = x[c]; s += v * v; }
  #pragma unroll
  for (int m = 16; m >= 1; m >>= 1) s += __shfl_xor(s, m, 32);
  __shared__ float red[8];
  __shared__ float sc_s;
  if ((threadIdx.x & 31) == 0) red[threadIdx.x >> 5] = s;
  __syncthreads();
  if (threadIdx.x == 0) {
    float t = 0.f;
    #pragma unroll
    for (int i = 0; i < 8; ++i) t += red[i];
    sc_s = rsqrtf(t / (float)ncols + 1e-6f);
  }
  __syncthreads();
  float sc = sc_s;
  __bf16* o = out + (size_t)row * ncols;
  for (int c = threadIdx.x; c < ncols; c += 256) o[c] = (__bf16)(x[c] * sc * w[c]);
}

// ------------------------------------------- kv split: RMSNorm(kv_c) + RoPE(k_pe)
__global__ __launch_bounds__(256) void kv_prep(const float* __restrict__ kv,
                                               const float* __restrict__ w,
                                               const float* __restrict__ freqs,
                                               __bf16* __restrict__ kvc,
                                               __bf16* __restrict__ kpe) {
  const int row = blockIdx.x;
  const int pos = row % S_;
  const float* x = kv + (size_t)row * (KVL_ + DR_);
  float s = 0.f;
  for (int c = threadIdx.x; c < KVL_; c += 256) { float v = x[c]; s += v * v; }
  #pragma unroll
  for (int m = 16; m >= 1; m >>= 1) s += __shfl_xor(s, m, 32);
  __shared__ float red[8];
  __shared__ float sc_s;
  if ((threadIdx.x & 31) == 0) red[threadIdx.x >> 5] = s;
  __syncthreads();
  if (threadIdx.x == 0) {
    float t = 0.f;
    #pragma unroll
    for (int i = 0; i < 8; ++i) t += red[i];
    sc_s = rsqrtf(t / (float)KVL_ + 1e-6f);
  }
  __syncthreads();
  float sc = sc_s;
  for (int c = threadIdx.x; c < KVL_; c += 256)
    kvc[(size_t)row * KVL_ + c] = (__bf16)(x[c] * sc * w[c]);
  if (threadIdx.x < DR_ / 2) {
    int p = threadIdx.x;
    float re = x[KVL_ + 2 * p], im = x[KVL_ + 2 * p + 1];
    float cs = freqs[((size_t)pos * 32 + p) * 2];
    float sn = freqs[((size_t)pos * 32 + p) * 2 + 1];
    kpe[(size_t)row * DR_ + 2 * p]     = (__bf16)(re * cs - im * sn);
    kpe[(size_t)row * DR_ + 2 * p + 1] = (__bf16)(re * sn + im * cs);
  }
}

// ---------------------------------------------------------------- RoPE on q
// Also folds the softmax scale into q so the attention hot loop has no
// post-WMMA scale multiply (avoids WMMA->VALU hazard stalls).
__global__ __launch_bounds__(256) void rope_q(const float* __restrict__ q,
                                              const float* __restrict__ freqs,
                                              __bf16* __restrict__ out) {
  const int row = blockIdx.x;               // B*S token rows
  const int pos = row % S_;
  const float* x = q + (size_t)row * (H_ * DQK_);
  __bf16* o = out + (size_t)row * (H_ * DQK_);
  for (int idx = threadIdx.x; idx < H_ * DQK_; idx += 256) {
    int d = idx % DQK_;
    if (d < DN_) {
      o[idx] = (__bf16)(x[idx] * SM_SCALE);
    } else if (((d - DN_) & 1) == 0) {
      int p = (d - DN_) >> 1;
      float re = x[idx], im = x[idx + 1];
      float cs = freqs[((size_t)pos * 32 + p) * 2];
      float sn = freqs[((size_t)pos * 32 + p) * 2 + 1];
      o[idx]     = (__bf16)((re * cs - im * sn) * SM_SCALE);
      o[idx + 1] = (__bf16)((re * sn + im * cs) * SM_SCALE);
    }
  }
}

// ---------------------------------------------------------------- flash attention
// q:   [B,S,H,192] bf16 (pre-scaled)  kvb: [B,S,H,256] bf16 (k_nope | v)
// kpe: [B,S,64] bf16                  out: [B,S,H*128] bf16
// 128 query rows per block (16 per wave), 64-key tiles.
#define KLD 200   // Ks stride (192 + 8)
#define VLD 72    // Vs / Ps stride (64 + 8)

__global__ __launch_bounds__(256) void mla_attn(const __bf16* __restrict__ q,
                                                const __bf16* __restrict__ kvb,
                                                const __bf16* __restrict__ kpe,
                                                __bf16* __restrict__ out) {
  const int h = blockIdx.y, b = blockIdx.z;
  const int q0 = blockIdx.x * 128;
  const int tid = threadIdx.x, wave = tid >> 5, lane = tid & 31;
  const int half = lane >> 4, ln = lane & 15;

  __shared__ __bf16 Ks[64 * KLD];            // [key][192]
  __shared__ __bf16 Vs[128 * VLD];           // transposed: [feature][key]
  __shared__ __bf16 Ps[8][16 * VLD];         // per-wave P spill

  // Q fragments: 16 rows x 192, straight from global in fragment layout.
  v16bf qf[6];
  {
    int qrow = q0 + wave * 16 + ln;
    const __bf16* qp = q + (((size_t)(b * S_ + qrow) * H_) + h) * DQK_ + (half << 3);
    #pragma unroll
    for (int kc = 0; kc < 6; ++kc) {
      v16bf f;
      ((uint4*)&f)[0] = *(const uint4*)(qp + kc * 32);
      ((uint4*)&f)[1] = *(const uint4*)(qp + kc * 32 + 16);
      qf[kc] = f;
    }
  }

  float m_run[8], l_run[8];
  #pragma unroll
  for (int r = 0; r < 8; ++r) { m_run[r] = -1e30f; l_run[r] = 0.f; }
  v8f oacc[8] = {};

  const int kv_hi = q0 + 128;                // causal bound for this block
  for (int kv0 = 0; kv0 < kv_hi; kv0 += 64) {
    // K tile: 64 x 192  (cols 0..127 from kvb nope, 128..191 from kpe)
    #pragma unroll
    for (int i = 0; i < 6; ++i) {
      int c = tid + i * 256;
      int r = c / 24, col = (c % 24) * 8;
      int key = kv0 + r;
      const __bf16* src = (col < DN_)
          ? kvb + (((size_t)(b * S_ + key) * H_) + h) * 256 + col
          : kpe + (size_t)(b * S_ + key) * DR_ + (col - DN_);
      *(uint4*)(&Ks[r * KLD + col]) = *(const uint4*)src;
    }
    // V tile transposed: Vs[f][key] = kvb[..][128+f]
    #pragma unroll
    for (int i = 0; i < 4; ++i) {
      int c = tid + i * 256;
      int r = c >> 4, f0 = (c & 15) << 3;
      int key = kv0 + r;
      uint4 v = *(const uint4*)(kvb + (((size_t)(b * S_ + key) * H_) + h) * 256 + DN_ + f0);
      const __bf16* e = (const __bf16*)&v;
      #pragma unroll
      for (int t = 0; t < 8; ++t) Vs[(f0 + t) * VLD + r] = e[t];
    }
    __syncthreads();

    // scores: 16 x 64 per wave; load fragment batch, then WMMA burst
    v8f sc[4];
    #pragma unroll
    for (int nt = 0; nt < 4; ++nt) {
      v16bf kf[6];
      #pragma unroll
      for (int kc = 0; kc < 6; ++kc) kf[kc] = ld_frag(Ks, nt * 16, kc * 32, KLD, lane);
      v8f a = {};
      #pragma unroll
      for (int kc = 0; kc < 6; ++kc) a = wmma_bf16(qf[kc], kf[kc], a);
      sc[nt] = a;
    }

    // online softmax (rows live in C-layout: row = r + 8*half, cols across 16 lanes)
    #pragma unroll
    for (int r = 0; r < 8; ++r) {
      int qi = q0 + wave * 16 + half * 8 + r;
      float mx = -1e30f;
      #pragma unroll
      for (int nt = 0; nt < 4; ++nt) {
        int kj = kv0 + nt * 16 + ln;
        float v = sc[nt][r];
        v = (kj <= qi) ? v : -1e30f;
        sc[nt][r] = v;
        mx = fmaxf(mx, v);
      }
      #pragma unroll
      for (int m = 8; m >= 1; m >>= 1) mx = fmaxf(mx, __shfl_xor(mx, m, 32));
      float mnew = fmaxf(m_run[r], mx);
      float alpha = __expf(m_run[r] - mnew);
      float rowsum = 0.f;
      #pragma unroll
      for (int nt = 0; nt < 4; ++nt) {
        float p = __expf(sc[nt][r] - mnew);
        sc[nt][r] = p;
        rowsum += p;
      }
      #pragma unroll
      for (int m = 8; m >= 1; m >>= 1) rowsum += __shfl_xor(rowsum, m, 32);
      l_run[r] = l_run[r] * alpha + rowsum;
      m_run[r] = mnew;
      #pragma unroll
      for (int o = 0; o < 8; ++o) oacc[o][r] *= alpha;
    }

    // spill P to per-wave LDS, reload as A fragments for P@V
    __bf16* pp = &Ps[wave][0];
    #pragma unroll
    for (int nt = 0; nt < 4; ++nt)
      #pragma unroll
      for (int r = 0; r < 8; ++r)
        pp[(half * 8 + r) * VLD + nt * 16 + ln] = (__bf16)sc[nt][r];

    #pragma unroll
    for (int kc = 0; kc < 2; ++kc) {
      v16bf pf = ld_frag(pp, 0, kc * 32, VLD, lane);
      v16bf vf[8];
      #pragma unroll
      for (int nt = 0; nt < 8; ++nt) vf[nt] = ld_frag(Vs, nt * 16, kc * 32, VLD, lane);
      #pragma unroll
      for (int nt = 0; nt < 8; ++nt) oacc[nt] = wmma_bf16(pf, vf[nt], oacc[nt]);
    }
    __syncthreads();
  }

  // epilogue: O * (1/l)  -> out[b, qi, h*128 + f]  (bf16)
  float inv_l[8];
  #pragma unroll
  for (int r = 0; r < 8; ++r) inv_l[r] = 1.0f / l_run[r];
  #pragma unroll
  for (int nt = 0; nt < 8; ++nt)
    #pragma unroll
    for (int r = 0; r < 8; ++r) {
      int qi = q0 + wave * 16 + half * 8 + r;
      int f  = nt * 16 + ln;
      out[((size_t)(b * S_ + qi)) * (H_ * DV_) + h * DV_ + f] =
          (__bf16)(oacc[nt][r] * inv_l[r]);
    }
}

// ---------------------------------------------------------------- launcher
extern "C" void kernel_launch(void* const* d_in, const int* in_sizes, int n_in,
                              void* d_out, int out_size, void* d_ws, size_t ws_size,
                              hipStream_t stream) {
  (void)in_sizes; (void)n_in; (void)out_size; (void)ws_size;
  const float* x     = (const float*)d_in[0];
  const float* freqs = (const float*)d_in[2];
  const float* wq_a  = (const float*)d_in[3];
  const float* q_nw  = (const float*)d_in[4];
  const float* wq_b  = (const float*)d_in[5];
  const float* wkv_a = (const float*)d_in[6];
  const float* kv_nw = (const float*)d_in[7];
  const float* wkv_b = (const float*)d_in[8];
  const float* wo    = (const float*)d_in[9];
  float* out = (float*)d_out;

  char* ws = (char*)d_ws;
  size_t off = 0;
  auto alloc = [&](size_t bytes) -> void* {
    void* p = ws + off;
    off = (off + bytes + 255) & ~(size_t)255;
    return p;
  };

  __bf16* x_bf     = (__bf16*)alloc((size_t)NTOK * D_ * 2);
  __bf16* wq_a_bf  = (__bf16*)alloc((size_t)QL_ * D_ * 2);
  __bf16* wq_b_bf  = (__bf16*)alloc((size_t)H_ * DQK_ * QL_ * 2);
  __bf16* wkv_a_bf = (__bf16*)alloc((size_t)(KVL_ + DR_) * D_ * 2);
  __bf16* wkv_b_bf = (__bf16*)alloc((size_t)H_ * (DN_ + DV_) * KVL_ * 2);
  __bf16* wo_bf    = (__bf16*)alloc((size_t)D_ * H_ * DV_ * 2);
  float*  q_a_f    = (float*)alloc((size_t)NTOK * QL_ * 4);
  __bf16* q_a_n    = (__bf16*)alloc((size_t)NTOK * QL_ * 2);
  float*  q_f      = (float*)alloc((size_t)NTOK * H_ * DQK_ * 4);
  __bf16* q_bf     = (__bf16*)alloc((size_t)NTOK * H_ * DQK_ * 2);
  float*  kv_f     = (float*)alloc((size_t)NTOK * (KVL_ + DR_) * 4);
  __bf16* kvc_bf   = (__bf16*)alloc((size_t)NTOK * KVL_ * 2);
  __bf16* kpe_bf   = (__bf16*)alloc((size_t)NTOK * DR_ * 2);
  __bf16* kvb_bf   = (__bf16*)alloc((size_t)NTOK * H_ * (DN_ + DV_) * 2);
  __bf16* attn_bf  = (__bf16*)alloc((size_t)NTOK * H_ * DV_ * 2);

  // fp32 -> bf16 conversions
  cvt_f32_bf16<<<1024, 256, 0, stream>>>(x,     x_bf,     (size_t)NTOK * D_);
  cvt_f32_bf16<<<1024, 256, 0, stream>>>(wq_a,  wq_a_bf,  (size_t)QL_ * D_);
  cvt_f32_bf16<<<1024, 256, 0, stream>>>(wq_b,  wq_b_bf,  (size_t)H_ * DQK_ * QL_);
  cvt_f32_bf16<<<1024, 256, 0, stream>>>(wkv_a, wkv_a_bf, (size_t)(KVL_ + DR_) * D_);
  cvt_f32_bf16<<<1024, 256, 0, stream>>>(wkv_b, wkv_b_bf, (size_t)H_ * (DN_ + DV_) * KVL_);
  cvt_f32_bf16<<<1024, 256, 0, stream>>>(wo,    wo_bf,    (size_t)D_ * H_ * DV_);

  dim3 blk(256);
  // q_a = x @ wq_a^T
  gemm_nt<false><<<dim3(QL_ / BN, NTOK / BM), blk, 0, stream>>>(
      x_bf, wq_a_bf, q_a_f, nullptr, NTOK, QL_, D_);
  rmsnorm_rows<<<NTOK, blk, 0, stream>>>(q_a_f, q_nw, q_a_n, QL_);
  // q = rmsnorm(q_a) @ wq_b^T
  gemm_nt<false><<<dim3(H_ * DQK_ / BN, NTOK / BM), blk, 0, stream>>>(
      q_a_n, wq_b_bf, q_f, nullptr, NTOK, H_ * DQK_, QL_);
  rope_q<<<NTOK, blk, 0, stream>>>(q_f, freqs, q_bf);
  // kv = x @ wkv_a^T
  gemm_nt<false><<<dim3((KVL_ + DR_ + BN - 1) / BN, NTOK / BM), blk, 0, stream>>>(
      x_bf, wkv_a_bf, kv_f, nullptr, NTOK, KVL_ + DR_, D_);
  kv_prep<<<NTOK, blk, 0, stream>>>(kv_f, kv_nw, freqs, kvc_bf, kpe_bf);
  // kvb = rmsnorm(kv_c) @ wkv_b^T  (bf16 out, consumed directly by attention)
  gemm_nt<true><<<dim3(H_ * (DN_ + DV_) / BN, NTOK / BM), blk, 0, stream>>>(
      kvc_bf, wkv_b_bf, nullptr, kvb_bf, NTOK, H_ * (DN_ + DV_), KVL_);
  // causal flash attention
  mla_attn<<<dim3(S_ / 128, H_, B_), blk, 0, stream>>>(q_bf, kvb_bf, kpe_bf, attn_bf);
  // out = attn @ wo^T (fp32 out)
  gemm_nt<false><<<dim3(D_ / BN, NTOK / BM), blk, 0, stream>>>(
      attn_bf, wo_bf, out, nullptr, NTOK, D_, H_ * DV_);
}
